// SimpleLLM_25039659336304
// MI455X (gfx1250) — compile-verified, matrix operand order checked by reading, not run on hardware
//
#include <hip/hip_runtime.h>
#include <hip/hip_bf16.h>
#include <math.h>

typedef _Float16 f16;
typedef __attribute__((ext_vector_type(16))) _Float16 v16h;
typedef __attribute__((ext_vector_type(8)))  _Float16 v8h;
typedef __attribute__((ext_vector_type(4)))  _Float16 v4h;
typedef __attribute__((ext_vector_type(8)))  float    v8f;

static __device__ __forceinline__ v16h cat16(v8h a, v8h b) {
    return __builtin_shufflevector(a, b, 0,1,2,3,4,5,6,7,8,9,10,11,12,13,14,15);
}

// ---------------------------------------------------------------------------
// Tensor Data Mover support (gfx1250). 2D tile load Global->LDS with LDS row
// padding so a 64x32 f16 tile lands with a 40-f16 (80B) row stride.
// ---------------------------------------------------------------------------
#if defined(__has_builtin)
#if __has_builtin(__builtin_amdgcn_tensor_load_to_lds)
#define USE_TDM 1
#endif
#endif
#ifndef USE_TDM
#define USE_TDM 0
#endif

#if USE_TDM
typedef unsigned int u32x4 __attribute__((ext_vector_type(4)));
typedef int          i32x4 __attribute__((ext_vector_type(4)));
typedef int          i32x8 __attribute__((ext_vector_type(8)));

// Issue TENSOR_LOAD_TO_LDS for a 64-row x 32-f16 tile starting at gsrc,
// global row stride = strideElems f16, into LDS byte address ldsaddr.
// D# group0/group1 packed per CDNA5 ISA/S8.4.
static __device__ __forceinline__ void tdm_load_tile_a(const f16* gsrc,
                                                       unsigned int ldsaddr,
                                                       int strideElems) {
    unsigned long long ga = (unsigned long long)(uintptr_t)gsrc;
    u32x4 g0;
    g0[0] = 1u;                                     // count=1 (valid user D#)
    g0[1] = ldsaddr;                                // lds_addr (bytes)
    g0[2] = (unsigned int)(ga & 0xFFFFFFFFu);       // global_addr[31:0]
    g0[3] = (unsigned int)((ga >> 32) & 0x01FFFFFFu) | (2u << 30); // addr[56:32], type=2

    unsigned int w0 = (1u << 16)    // data_size = 2 bytes
                    | (1u << 20)    // pad_enable
                    | (3u << 22)    // pad_interval: every 16 DWORDs (= one 64B row)
                    | (3u << 25);   // pad_amount: 4 DWORDs (8 f16) -> 40-f16 stride
    unsigned int td0 = 0x40000000u, td1 = 0x40000000u; // huge dims: no OOB clip
    unsigned int tile0 = 32u, tile1 = 64u;
    unsigned long long s0 = (unsigned long long)(unsigned int)strideElems;

    i32x8 g1;
    g1[0] = (int)w0;
    g1[1] = (int)((td0 & 0xFFFFu) << 16);                  // abar=0 | td0[15:0]
    g1[2] = (int)((td0 >> 16) | ((td1 & 0xFFFFu) << 16));  // td0[31:16] | td1[15:0]
    g1[3] = (int)((td1 >> 16) | (tile0 << 16));            // td1[31:16] | tile_dim0
    g1[4] = (int)tile1;                                    // tile_dim1 | tile_dim2=0
    g1[5] = (int)(unsigned int)(s0 & 0xFFFFFFFFu);         // dim0_stride lo
    g1[6] = (int)(unsigned int)((s0 >> 32) & 0xFFFFu);     // dim0_stride hi | s1 lo=0
    g1[7] = 0;
    i32x4 z4 = {0, 0, 0, 0};
#if __clang_major__ >= 23
    i32x8 z8 = {0, 0, 0, 0, 0, 0, 0, 0};
    __builtin_amdgcn_tensor_load_to_lds(g0, g1, z4, z4, z8, 0);
#else
    __builtin_amdgcn_tensor_load_to_lds(g0, g1, z4, z4, 0);
#endif
}
#endif // USE_TDM

// ---------------------------------------------------------------------------
// Embedding gather: x[t, :] = embed[ids[t], :]
// ---------------------------------------------------------------------------
__global__ __launch_bounds__(256) void embed_kernel(const int* __restrict__ ids,
                                                    const float* __restrict__ emb,
                                                    float* __restrict__ x) {
    const int D = 2048;
    int t = blockIdx.x;
    int id = ids[t];
    const float4* src = (const float4*)(emb + (size_t)id * D);
    float4* dst = (float4*)(x + (size_t)t * D);
    for (int i = threadIdx.x; i < D / 4; i += 256) dst[i] = src[i];
}

// ---------------------------------------------------------------------------
// RMSNorm: out_f16[t, :] = x[t, :] * rsqrt(mean(x^2) + eps) * w
// ---------------------------------------------------------------------------
__global__ __launch_bounds__(256) void rmsnorm_kernel(const float* __restrict__ x,
                                                      const float* __restrict__ w,
                                                      f16* __restrict__ out) {
    const int D = 2048;
    int t = blockIdx.x;
    const float* xr = x + (size_t)t * D;
    float ss = 0.f;
    for (int i = threadIdx.x; i < D; i += 256) { float v = xr[i]; ss += v * v; }
    ss += __shfl_xor(ss, 1, 32);
    ss += __shfl_xor(ss, 2, 32);
    ss += __shfl_xor(ss, 4, 32);
    ss += __shfl_xor(ss, 8, 32);
    ss += __shfl_xor(ss, 16, 32);
    __shared__ float red[8];
    __shared__ float rs_sh;
    if ((threadIdx.x & 31) == 0) red[threadIdx.x >> 5] = ss;
    __syncthreads();
    if (threadIdx.x == 0) {
        float s = 0.f;
        for (int j = 0; j < 8; j++) s += red[j];
        rs_sh = rsqrtf(s / (float)D + 1e-6f);
    }
    __syncthreads();
    float rs = rs_sh;
    for (int i = threadIdx.x; i < D; i += 256)
        out[(size_t)t * D + i] = (f16)(xr[i] * rs * w[i]);
}

// ---------------------------------------------------------------------------
// RoPE for q,k + f16 repack of q,k,v into [B, H, S, hd] layouts
// ---------------------------------------------------------------------------
__global__ __launch_bounds__(256) void rope_pack_kernel(const float* __restrict__ q,
                                                        const float* __restrict__ k,
                                                        const float* __restrict__ v,
                                                        f16* __restrict__ qh,
                                                        f16* __restrict__ kh,
                                                        f16* __restrict__ vh) {
    const int D = 2048, KVD = 1024, S = 1024, HD = 128;
    int t = blockIdx.x;
    int b = t >> 10;
    int s = t & 1023;
    const float LOG_THETA_OVER_HALF = 0.14391156645f; // ln(10000)/64

    for (int idx = threadIdx.x; idx < 1024; idx += 256) {
        int h = idx >> 6, i = idx & 63;
        float inv = __expf(-(float)i * LOG_THETA_OVER_HALF);
        float ang = (float)s * inv;
        float c = __cosf(ang), sn = __sinf(ang);
        float x1 = q[(size_t)t * D + h * HD + i];
        float x2 = q[(size_t)t * D + h * HD + 64 + i];
        size_t o = ((size_t)(b * 16 + h) * S + s) * HD;
        qh[o + i]      = (f16)(x1 * c - x2 * sn);
        qh[o + 64 + i] = (f16)(x1 * sn + x2 * c);
    }
    for (int idx = threadIdx.x; idx < 512; idx += 256) {
        int h = idx >> 6, i = idx & 63;
        float inv = __expf(-(float)i * LOG_THETA_OVER_HALF);
        float ang = (float)s * inv;
        float c = __cosf(ang), sn = __sinf(ang);
        float x1 = k[(size_t)t * KVD + h * HD + i];
        float x2 = k[(size_t)t * KVD + h * HD + 64 + i];
        size_t o = ((size_t)(b * 8 + h) * S + s) * HD;
        kh[o + i]      = (f16)(x1 * c - x2 * sn);
        kh[o + 64 + i] = (f16)(x1 * sn + x2 * c);
    }
    for (int idx = threadIdx.x; idx < 1024; idx += 256) {
        int h = idx >> 7, d = idx & 127;
        vh[((size_t)(b * 8 + h) * S + s) * HD + d] = (f16)v[(size_t)t * KVD + h * HD + d];
    }
}

// ---------------------------------------------------------------------------
// Tiled WMMA GEMM: C[M,N] = A_f16[M,K] * W_f32[N,K]^T   (W converted to f16)
// MODE 0: C = acc   MODE 1: C += acc   MODE 2: C16 = (f16)(silu(acc)*aux)
// Block 256 threads (8 waves, 2x4); block tile 64x128; wave tile 32x32
// (2x2 WMMA register block). Double-buffered LDS; A tile staged by the
// Tensor Data Mover (wave 0 issues, s_wait_tensorcnt fences), W tile staged
// by vector loads + f32->f16 convert.
// ---------------------------------------------------------------------------
template <int MODE>
__global__ __launch_bounds__(256) void gemm_wmma(const f16* __restrict__ A,
                                                 const float* __restrict__ W,
                                                 float* __restrict__ C,
                                                 f16* __restrict__ C16,
                                                 const float* __restrict__ aux,
                                                 int M, int N, int K) {
    __shared__ f16 As[2][64][40];   // 80B row stride (16B aligned, TDM pad layout)
    __shared__ f16 Bs[2][128][40];

    int tid = threadIdx.x;
    int bm = blockIdx.y * 64;
    int bn = blockIdx.x * 128;
    int w = tid >> 5, lane = tid & 31;
    int wm = (w & 1) * 32, wn = (w >> 1) * 32;
    int r = lane & 15, lh = lane >> 4, kb = lh * 8;

    int brow = tid >> 1, bcol = (tid & 1) * 16;   // 128 rows x 32 f32 (16/thread)
#if !USE_TDM
    int arow = tid >> 2, acol = (tid & 3) * 8;    // 64 rows x 32 f16 (8/thread)
#endif

    v8f acc[2][2];
#pragma unroll
    for (int a = 0; a < 2; a++)
#pragma unroll
        for (int b2 = 0; b2 < 2; b2++)
            acc[a][b2] = (v8f){0.f,0.f,0.f,0.f,0.f,0.f,0.f,0.f};

    auto stageA = [&](int buf, int k0) {
#if USE_TDM
        if (tid < 32)
            tdm_load_tile_a(&A[(size_t)bm * K + k0],
                            (unsigned int)(uintptr_t)&As[buf][0][0], K);
#else
        *(v8h*)&As[buf][arow][acol] =
            *(const v8h*)&A[(size_t)(bm + arow) * K + k0 + acol];
#endif
    };
    auto stageB = [&](int buf, int k0) {
        const float* wp = &W[(size_t)(bn + brow) * K + k0 + bcol];
        float4 f0 = *(const float4*)(wp);
        float4 f1 = *(const float4*)(wp + 4);
        float4 f2 = *(const float4*)(wp + 8);
        float4 f3 = *(const float4*)(wp + 12);
        if (k0 + 32 < K)
            __builtin_prefetch(wp + 32, 0, 0);
        v8h h0, h1;
        h0[0] = (f16)f0.x; h0[1] = (f16)f0.y; h0[2] = (f16)f0.z; h0[3] = (f16)f0.w;
        h0[4] = (f16)f1.x; h0[5] = (f16)f1.y; h0[6] = (f16)f1.z; h0[7] = (f16)f1.w;
        h1[0] = (f16)f2.x; h1[1] = (f16)f2.y; h1[2] = (f16)f2.z; h1[3] = (f16)f2.w;
        h1[4] = (f16)f3.x; h1[5] = (f16)f3.y; h1[6] = (f16)f3.z; h1[7] = (f16)f3.w;
        *(v8h*)&Bs[buf][brow][bcol]     = h0;
        *(v8h*)&Bs[buf][brow][bcol + 8] = h1;
    };

    stageA(0, 0);
    stageB(0, 0);
#if USE_TDM
    if (tid < 32) __builtin_amdgcn_s_wait_tensorcnt(0);
#endif
    __syncthreads();

    int steps = K >> 5;
    int buf = 0;
    for (int s = 0; s < steps; s++) {
        if (s + 1 < steps) {
            stageA(buf ^ 1, (s + 1) << 5);
            stageB(buf ^ 1, (s + 1) << 5);
        }

        v16h a0 = cat16(*(const v8h*)&As[buf][wm + r][kb],
                        *(const v8h*)&As[buf][wm + r][kb + 16]);
        v16h a1 = cat16(*(const v8h*)&As[buf][wm + 16 + r][kb],
                        *(const v8h*)&As[buf][wm + 16 + r][kb + 16]);
        v16h b0 = cat16(*(const v8h*)&Bs[buf][wn + r][kb],
                        *(const v8h*)&Bs[buf][wn + r][kb + 16]);
        v16h b1 = cat16(*(const v8h*)&Bs[buf][wn + 16 + r][kb],
                        *(const v8h*)&Bs[buf][wn + 16 + r][kb + 16]);

        acc[0][0] = __builtin_amdgcn_wmma_f32_16x16x32_f16(false, a0, false, b0,
                                                           (short)0, acc[0][0], false, false);
        acc[0][1] = __builtin_amdgcn_wmma_f32_16x16x32_f16(false, a0, false, b1,
                                                           (short)0, acc[0][1], false, false);
        acc[1][0] = __builtin_amdgcn_wmma_f32_16x16x32_f16(false, a1, false, b0,
                                                           (short)0, acc[1][0], false, false);
        acc[1][1] = __builtin_amdgcn_wmma_f32_16x16x32_f16(false, a1, false, b1,
                                                           (short)0, acc[1][1], false, false);

        if (s + 1 < steps) {
#if USE_TDM
            if (tid < 32) __builtin_amdgcn_s_wait_tensorcnt(0);
#endif
            __syncthreads();
            buf ^= 1;
        }
    }

#pragma unroll
    for (int tm = 0; tm < 2; tm++) {
#pragma unroll
        for (int tn = 0; tn < 2; tn++) {
#pragma unroll
            for (int i = 0; i < 8; i++) {
                int row = bm + wm + tm * 16 + lh * 8 + i;
                int col = bn + wn + tn * 16 + r;
                size_t idx = (size_t)row * N + col;
                float vv = acc[tm][tn][i];
                if (MODE == 0) {
                    C[idx] = vv;
                } else if (MODE == 1) {
                    C[idx] += vv;
                } else { // silu(gate) * up -> f16
                    float sig = 1.f / (1.f + __expf(-vv));
                    C16[idx] = (f16)(vv * sig * aux[idx]);
                }
            }
        }
    }
}

// ---------------------------------------------------------------------------
// Flash attention with WMMA. One block per (b, head, 128-row q tile).
// ---------------------------------------------------------------------------
__global__ __launch_bounds__(256) void attn_kernel(const f16* __restrict__ qh,
                                                   const f16* __restrict__ kh,
                                                   const f16* __restrict__ vh,
                                                   f16* __restrict__ ah) {
    const int HD = 128, S = 1024;
    int bx = blockIdx.x;
    int qt = bx & 7;
    int h = (bx >> 3) & 15;
    int b = bx >> 7;
    int kvh = h >> 1;

    const f16* Q  = qh + (size_t)(b * 16 + h)   * S * HD;
    const f16* Kg = kh + (size_t)(b * 8 + kvh) * S * HD;
    const f16* Vg = vh + (size_t)(b * 8 + kvh) * S * HD;

    __shared__ f16 Ks[32][HD + 8];
    __shared__ f16 Vts[HD][40];
    __shared__ f16 Pb[8][16 * 32];

    int tid = threadIdx.x;
    int w = tid >> 5, lane = tid & 31;
    int r = lane & 15, lh = lane >> 4, kb = lh * 8;
    int qbase = qt * 128 + w * 16;

    v16h af[4];
#pragma unroll
    for (int c = 0; c < 4; c++) {
        const f16* qp = Q + (size_t)(qbase + r) * HD + c * 32 + kb;
        af[c] = cat16(*(const v8h*)qp, *(const v8h*)(qp + 16));
    }

    v8f o[8];
#pragma unroll
    for (int c = 0; c < 8; c++) o[c] = (v8f){0.f,0.f,0.f,0.f,0.f,0.f,0.f,0.f};
    float mrow[8], lrow[8];
#pragma unroll
    for (int i = 0; i < 8; i++) { mrow[i] = -1e30f; lrow[i] = 0.f; }

    const float sc = 0.08838834764831845f;
    int nkb = (qt + 1) * 4;

    for (int kblk = 0; kblk < nkb; kblk++) {
        int kpos0 = kblk * 32;
        {
            int row = tid >> 3, cg = (tid & 7) * 16;
            const f16* kp = &Kg[(size_t)(kpos0 + row) * HD + cg];
            *(v8h*)&Ks[row][cg]     = *(const v8h*)kp;
            *(v8h*)&Ks[row][cg + 8] = *(const v8h*)(kp + 8);
            const f16* vp = &Vg[(size_t)(kpos0 + row) * HD + cg];
            v8h v0 = *(const v8h*)vp;
            v8h v1 = *(const v8h*)(vp + 8);
#pragma unroll
            for (int j = 0; j < 8; j++) {
                Vts[cg + j][row]     = v0[j];
                Vts[cg + 8 + j][row] = v1[j];
            }
        }
        __syncthreads();

        v8f s0 = {0.f,0.f,0.f,0.f,0.f,0.f,0.f,0.f};
        v8f s1 = {0.f,0.f,0.f,0.f,0.f,0.f,0.f,0.f};
#pragma unroll
        for (int c = 0; c < 4; c++) {
            v16h b0 = cat16(*(const v8h*)&Ks[r][c * 32 + kb],
                            *(const v8h*)&Ks[r][c * 32 + kb + 16]);
            v16h b1 = cat16(*(const v8h*)&Ks[16 + r][c * 32 + kb],
                            *(const v8h*)&Ks[16 + r][c * 32 + kb + 16]);
            s0 = __builtin_amdgcn_wmma_f32_16x16x32_f16(false, af[c], false, b0,
                                                        (short)0, s0, false, false);
            s1 = __builtin_amdgcn_wmma_f32_16x16x32_f16(false, af[c], false, b1,
                                                        (short)0, s1, false, false);
        }

        f16* pb = &Pb[w][0];
#pragma unroll
        for (int i = 0; i < 8; i++) {
            int qrow = qbase + lh * 8 + i;
            float s0v = (kpos0 + r      <= qrow) ? s0[i] * sc : -1e30f;
            float s1v = (kpos0 + 16 + r <= qrow) ? s1[i] * sc : -1e30f;
            float mx = fmaxf(s0v, s1v);
            mx = fmaxf(mx, __shfl_xor(mx, 1, 32));
            mx = fmaxf(mx, __shfl_xor(mx, 2, 32));
            mx = fmaxf(mx, __shfl_xor(mx, 4, 32));
            mx = fmaxf(mx, __shfl_xor(mx, 8, 32));
            float mnew = fmaxf(mrow[i], mx);
            float corr = __expf(mrow[i] - mnew);
            float p0 = __expf(s0v - mnew);
            float p1 = __expf(s1v - mnew);
            float rs2 = p0 + p1;
            rs2 += __shfl_xor(rs2, 1, 32);
            rs2 += __shfl_xor(rs2, 2, 32);
            rs2 += __shfl_xor(rs2, 4, 32);
            rs2 += __shfl_xor(rs2, 8, 32);
            lrow[i] = lrow[i] * corr + rs2;
            mrow[i] = mnew;
#pragma unroll
            for (int c = 0; c < 8; c++) o[c][i] *= corr;
            pb[(lh * 8 + i) * 32 + r]      = (f16)p0;
            pb[(lh * 8 + i) * 32 + 16 + r] = (f16)p1;
        }

        v16h pf = cat16(*(const v8h*)&pb[r * 32 + kb],
                        *(const v8h*)&pb[r * 32 + kb + 16]);

#pragma unroll
        for (int c = 0; c < 8; c++) {
            v16h bv = cat16(*(const v8h*)&Vts[c * 16 + r][kb],
                            *(const v8h*)&Vts[c * 16 + r][kb + 16]);
            o[c] = __builtin_amdgcn_wmma_f32_16x16x32_f16(false, pf, false, bv,
                                                          (short)0, o[c], false, false);
        }
        __syncthreads();
    }

#pragma unroll
    for (int c = 0; c < 8; c++) {
#pragma unroll
        for (int i = 0; i < 8; i++) {
            int row = qbase + lh * 8 + i;
            size_t tg = (size_t)b * S + row;
            ah[tg * 2048 + h * 128 + c * 16 + r] = (f16)(o[c][i] / lrow[i]);
        }
    }
}

// ---------------------------------------------------------------------------
// Host orchestration
// ---------------------------------------------------------------------------
extern "C" void kernel_launch(void* const* d_in, const int* in_sizes, int n_in,
                              void* d_out, int out_size, void* d_ws, size_t ws_size,
                              hipStream_t stream) {
    (void)in_sizes; (void)n_in; (void)out_size; (void)ws_size;
    const int T = 2048, D = 2048, FF = 8192, KVD = 1024, VOC = 32000, NL = 4;

    const int*   ids    = (const int*)  d_in[0];
    const float* embed  = (const float*)d_in[3];
    const float* ln1    = (const float*)d_in[4];
    const float* wq     = (const float*)d_in[5];
    const float* wk     = (const float*)d_in[6];
    const float* wv     = (const float*)d_in[7];
    const float* wo     = (const float*)d_in[8];
    const float* ln2    = (const float*)d_in[9];
    const float* wg     = (const float*)d_in[10];
    const float* wu     = (const float*)d_in[11];
    const float* wd     = (const float*)d_in[12];
    const float* norm_w = (const float*)d_in[13];
    const float* lmh    = (const float*)d_in[14];

    char* ws = (char*)d_ws;
    size_t off = 0;
    auto take = [&](size_t bytes) -> void* {
        void* p = ws + off;
        off += (bytes + 255) & ~(size_t)255;
        return p;
    };
    float* x   = (float*)take((size_t)T * D * 4);
    f16*   xn  = (f16*)  take((size_t)T * D * 2);
    float* qf  = (float*)take((size_t)T * D * 4);
    float* kf  = (float*)take((size_t)T * KVD * 4);
    float* vf  = (float*)take((size_t)T * KVD * 4);
    f16*   q16 = (f16*)  take((size_t)T * D * 2);
    f16*   k16 = (f16*)  take((size_t)T * KVD * 2);
    f16*   v16 = (f16*)  take((size_t)T * KVD * 2);
    f16*   a16 = (f16*)  take((size_t)T * D * 2);
    float* uf  = (float*)take((size_t)T * FF * 4);
    f16*   h16 = (f16*)  take((size_t)T * FF * 2);

    dim3 blk(256);

    embed_kernel<<<T, blk, 0, stream>>>(ids, embed, x);

    for (int l = 0; l < NL; l++) {
        const float* lq = wq + (size_t)l * D * D;
        const float* lk = wk + (size_t)l * KVD * D;
        const float* lv = wv + (size_t)l * KVD * D;
        const float* lo = wo + (size_t)l * D * D;
        const float* lg = wg + (size_t)l * FF * D;
        const float* lu = wu + (size_t)l * FF * D;
        const float* ld = wd + (size_t)l * D * FF;

        rmsnorm_kernel<<<T, blk, 0, stream>>>(x, ln1 + (size_t)l * D, xn);

        gemm_wmma<0><<<dim3(D / 128,   T / 64), blk, 0, stream>>>(xn, lq, qf, nullptr, nullptr, T, D,   D);
        gemm_wmma<0><<<dim3(KVD / 128, T / 64), blk, 0, stream>>>(xn, lk, kf, nullptr, nullptr, T, KVD, D);
        gemm_wmma<0><<<dim3(KVD / 128, T / 64), blk, 0, stream>>>(xn, lv, vf, nullptr, nullptr, T, KVD, D);

        rope_pack_kernel<<<T, blk, 0, stream>>>(qf, kf, vf, q16, k16, v16);

        attn_kernel<<<256, blk, 0, stream>>>(q16, k16, v16, a16);

        gemm_wmma<1><<<dim3(D / 128, T / 64), blk, 0, stream>>>(a16, lo, x, nullptr, nullptr, T, D, D);

        rmsnorm_kernel<<<T, blk, 0, stream>>>(x, ln2 + (size_t)l * D, xn);

        gemm_wmma<0><<<dim3(FF / 128, T / 64), blk, 0, stream>>>(xn, lu, uf, nullptr, nullptr, T, FF, D);
        gemm_wmma<2><<<dim3(FF / 128, T / 64), blk, 0, stream>>>(xn, lg, nullptr, h16, uf, T, FF, D);

        gemm_wmma<1><<<dim3(D / 128, T / 64), blk, 0, stream>>>(h16, ld, x, nullptr, nullptr, T, D, FF);
    }

    rmsnorm_kernel<<<T, blk, 0, stream>>>(x, norm_w, xn);
    gemm_wmma<0><<<dim3(VOC / 128, T / 64), blk, 0, stream>>>(xn, lmh, (float*)d_out,
                                                              nullptr, nullptr, T, VOC, D);
}
